// DocREModel_50165218018108
// MI455X (gfx1250) — matrix operand order repaired and checked by reading, not compiled, not run on hardware
//
#include <hip/hip_runtime.h>

// ---------------- problem constants ----------------
constexpr int N_  = 4;
constexpr int C_  = 1024;
constexpr int D_  = 768;
constexpr int H_  = 12;
constexpr int E_  = 32;
constexpr int M_  = 4;
constexpr int P_  = 256;
constexpr int EMB_ = 768;
constexpr int BS_  = 64;
constexpr int NC_  = 97;
constexpr int OFFSET_ = 1;
constexpr int KTOT_ = EMB_ * BS_;   // 49152

// ---------------- WMMA types ----------------
typedef __bf16 v16bf __attribute__((ext_vector_type(16)));
typedef float  v8f   __attribute__((ext_vector_type(8)));
typedef unsigned int v4u __attribute__((ext_vector_type(4)));
typedef int v4i_t __attribute__((ext_vector_type(4)));

union Frag { v4u u[2]; v16bf v; };

// ---------------- CDNA5 async global->LDS probe ----------------
#if defined(__HIP_DEVICE_COMPILE__)
#if __has_builtin(__builtin_amdgcn_global_load_async_to_lds_b128) && \
    __has_builtin(__builtin_amdgcn_s_wait_asynccnt)
#define HAVE_ASYNC_LDS 1
#endif
#endif

#ifdef HAVE_ASYNC_LDS
typedef __attribute__((address_space(1))) v4i_t* gv4i_p;   // global int4*
typedef __attribute__((address_space(3))) v4i_t* lv4i_p;   // LDS int4*
__device__ __forceinline__ void async_cp16(const __bf16* g, __bf16* l) {
  // generic->AS pointer via integer round-trip (CK idiom):
  // global flat address == AS1 address; LDS flat address low 32 bits == LDS offset
  gv4i_p gp = reinterpret_cast<gv4i_p>((unsigned long long)g);
  lv4i_p lp = reinterpret_cast<lv4i_p>((unsigned int)(unsigned long long)l);
  __builtin_amdgcn_global_load_async_to_lds_b128(gp, lp, 0, 0);
}
#endif

// ---------------- helpers ----------------
__device__ __forceinline__ float block_reduce_sum(float v, float* sbuf) {
  int t = threadIdx.x;
  sbuf[t] = v;
  __syncthreads();
  for (int s = blockDim.x >> 1; s > 0; s >>= 1) {
    if (t < s) sbuf[t] += sbuf[t + s];
    __syncthreads();
  }
  float r = sbuf[0];
  __syncthreads();
  return r;
}

// ============================================================
// K1: per-(doc,entity) masked logsumexp + masked attention mean
// ============================================================
__global__ void entity_agg(const float* __restrict__ seq,
                           const float* __restrict__ att,
                           const int* __restrict__ m_idx,
                           const unsigned char* __restrict__ m_mask,
                           float* __restrict__ e_emb,
                           float* __restrict__ e_att) {
  int b = blockIdx.x;
  int n = b >> 5;
  int e = b & 31;
  int base = (n * E_ + e) * M_;

  int  pos[M_];
  bool val[M_];
  int  cnt = 0;
#pragma unroll
  for (int m = 0; m < M_; m++) {
    int idx = m_idx[base + m];
    int p = idx + OFFSET_;
    bool v = (m_mask[base + m] != 0) && (p < C_);
    p = (p < C_ - 1) ? p : (C_ - 1);
    pos[m] = p; val[m] = v;
    cnt += v ? 1 : 0;
  }
  const float* seqn = seq + (size_t)n * C_ * D_;
  float* ee = e_emb + (size_t)(n * E_ + e) * D_;
  for (int d = threadIdx.x; d < D_; d += blockDim.x) {
    float mx = -INFINITY;
#pragma unroll
    for (int m = 0; m < M_; m++) if (val[m]) mx = fmaxf(mx, seqn[(size_t)pos[m] * D_ + d]);
    float s = 0.f;
#pragma unroll
    for (int m = 0; m < M_; m++) if (val[m]) s += expf(seqn[(size_t)pos[m] * D_ + d] - mx);
    ee[d] = mx + logf(s);
  }
  float inv = 1.0f / (float)cnt;
  const float* attn = att + (size_t)n * H_ * C_ * C_;
  float* ea = e_att + (size_t)(n * E_ + e) * H_ * C_;
  for (int i = threadIdx.x; i < H_ * C_; i += blockDim.x) {
    int h = i >> 10;
    int c = i & (C_ - 1);
    float s = 0.f;
#pragma unroll
    for (int m = 0; m < M_; m++) if (val[m]) s += attn[((size_t)h * C_ + pos[m]) * C_ + c];
    ea[(size_t)h * C_ + c] = s * inv;
  }
}

// ============================================================
// K2: ht[n,p,c] = normalize_c( mean_h(h_att*t_att) )
// ============================================================
__global__ void pair_ht(const float* __restrict__ e_att,
                        const int* __restrict__ pairs,
                        float* __restrict__ ht) {
  __shared__ float row[C_];
  __shared__ float red[256];
  int b = blockIdx.x;
  int n = b >> 8;
  int p = b & 255;
  int he = pairs[(n * P_ + p) * 2 + 0];
  int te = pairs[(n * P_ + p) * 2 + 1];
  const float* ha = e_att + (size_t)(n * E_ + he) * H_ * C_;
  const float* ta = e_att + (size_t)(n * E_ + te) * H_ * C_;
  float lsum = 0.f;
  for (int c = threadIdx.x; c < C_; c += blockDim.x) {
    float acc = 0.f;
#pragma unroll
    for (int h = 0; h < H_; h++) acc += ha[h * C_ + c] * ta[h * C_ + c];
    float v = acc * (1.0f / H_);
    row[c] = v;
    lsum += v;
  }
  float S = block_reduce_sum(lsum, red);
  float inv = 1.0f / (S + 1e-5f);
  float* out = ht + (size_t)(n * P_ + p) * C_;
  for (int c = threadIdx.x; c < C_; c += blockDim.x) out[c] = row[c] * inv;
}

// ============================================================
// K2b: gather e_emb into concat buffers (cols 0..D-1)
// ============================================================
__global__ void gather_pairs(const float* __restrict__ e_emb,
                             const int* __restrict__ pairs,
                             float* __restrict__ catH,
                             float* __restrict__ catT) {
  int idx = blockIdx.x * blockDim.x + threadIdx.x;
  if (idx >= N_ * P_ * D_) return;
  int d = idx % D_;
  int row = idx / D_;
  int n = row / P_, p = row % P_;
  int he = pairs[(n * P_ + p) * 2 + 0];
  int te = pairs[(n * P_ + p) * 2 + 1];
  catH[(size_t)row * (2 * D_) + d] = e_emb[(size_t)(n * E_ + he) * D_ + d];
  catT[(size_t)row * (2 * D_) + d] = e_emb[(size_t)(n * E_ + te) * D_ + d];
}

// ============================================================
// Generic tiled GEMM via v_wmma_f32_16x16x32_bf16 (f32 in/out).
// ============================================================
template <bool TANH, bool OUTBF16>
__global__ void gemm_wmma(const float* __restrict__ A,
                          const float* __restrict__ B,
                          const float* __restrict__ bias,
                          void* __restrict__ C0, void* __restrict__ C1,
                          int Mdim, int Ndim, int Kdim,
                          int lda, int ldb, int ldc,
                          long sA, long sB, long sC) {
  extern __shared__ char smem[];
  __bf16* a_sh = (__bf16*)smem;        // 64 x 40
  __bf16* b_sh = a_sh + 64 * 40;       // 64 x 40 (transposed: [n][k])
  (void)Mdim; (void)Ndim;

  int bz = blockIdx.z;
  A += (size_t)bz * sA;
  B += (size_t)bz * sB;
  int row0 = blockIdx.x * 64, col0 = blockIdx.y * 64;
  int t = threadIdx.x;
  int wid = t >> 5, lane = t & 31, half = (lane >> 4) & 1, lr = lane & 15;
  int wm = wid >> 2, wn = wid & 3;

  v8f acc0 = {}, acc1 = {};

  for (int kb = 0; kb < Kdim; kb += 32) {
    {
      int k = t & 31, r0 = t >> 5;
#pragma unroll
      for (int s = 0; s < 8; s++) {
        int r = r0 + 8 * s;
        a_sh[r * 40 + k] = (__bf16)A[(size_t)(row0 + r) * lda + kb + k];
      }
      int c = t & 63, k0 = t >> 6;
#pragma unroll
      for (int s = 0; s < 8; s++) {
        int k2 = k0 + 4 * s;
        b_sh[c * 40 + k2] = (__bf16)B[(size_t)(kb + k2) * ldb + col0 + c];
      }
    }
    __syncthreads();

    Frag fa0, fa1, fb;
    const v4u* ar0 = (const v4u*)(a_sh + (wm * 32 + lr) * 40);
    const v4u* ar1 = (const v4u*)(a_sh + (wm * 32 + 16 + lr) * 40);
    const v4u* br  = (const v4u*)(b_sh + (wn * 16 + lr) * 40);
    fa0.u[0] = ar0[half];     fa0.u[1] = ar0[2 + half];
    fa1.u[0] = ar1[half];     fa1.u[1] = ar1[2 + half];
    fb.u[0]  = br[half];  fb.u[1]  = br[2 + half];

    acc0 = __builtin_amdgcn_wmma_f32_16x16x32_bf16(false, fa0.v, false, fb.v,
                                                   (short)0, acc0, false, false);
    acc1 = __builtin_amdgcn_wmma_f32_16x16x32_bf16(false, fa1.v, false, fb.v,
                                                   (short)0, acc1, false, false);
    __syncthreads();
  }

  int col = col0 + wn * 16 + lr;
  float bv = bias ? bias[col] : 0.0f;
#pragma unroll
  for (int v = 0; v < 8; v++) {
    int r = row0 + wm * 32 + v + 8 * half;
    float x0 = acc0[v] + bv;
    float x1 = acc1[v] + bv;
    if (TANH) { x0 = tanhf(x0); x1 = tanhf(x1); }
    size_t i0 = (size_t)bz * sC + (size_t)r * ldc + col;
    size_t i1 = i0 + (size_t)16 * ldc;
    if (OUTBF16) {
      ((__bf16*)C0)[i0] = (__bf16)x0;
      ((__bf16*)C0)[i1] = (__bf16)x1;
      if (C1) { ((__bf16*)C1)[i0] = (__bf16)x0; ((__bf16*)C1)[i1] = (__bf16)x1; }
    } else {
      ((float*)C0)[i0] = x0;
      ((float*)C0)[i1] = x1;
      if (C1) { ((float*)C1)[i0] = x0; ((float*)C1)[i1] = x1; }
    }
  }
}

// ============================================================
// Pre-pass: Wt[c][k] = (bf16) bil_W[k][c]   (64x64 tiles via LDS)
// grid (KTOT_/64 = 768, EMB_/64 = 12), 256 threads
// ============================================================
__global__ void conv_transpose_W(const float* __restrict__ W,
                                 __bf16* __restrict__ Wt) {
  __shared__ __bf16 tile[64][72];
  int kb = blockIdx.x * 64;
  int cb = blockIdx.y * 64;
  int t = threadIdx.x;
  int c = t & 63, k0 = t >> 6;
#pragma unroll
  for (int s = 0; s < 16; s++) {
    int k = k0 + 4 * s;
    tile[k][c] = (__bf16)W[(size_t)(kb + k) * EMB_ + cb + c];
  }
  __syncthreads();
  int k2 = t & 63, c0 = t >> 6;
#pragma unroll
  for (int s = 0; s < 16; s++) {
    int c2 = c0 + 4 * s;
    Wt[(size_t)(cb + c2) * KTOT_ + kb + k2] = tile[k2][c2];
  }
}

// ============================================================
// K5 (fast): bilinear GEMM with pre-transposed bf16 W.
//   A generated on the fly: A[r,k]=hs[r,g*64+i]*ts[r,g*64+j], k=g*4096+i*64+j
// K staged in 128-chunks, W slab double-buffered via async global->LDS
// copies (ASYNCcnt) when available; reg-staged b128 copies otherwise.
// LDS: ts 64x72 | hs 64x8 | W 2 x 64x136  => 45056 B
// ============================================================
__global__ void bilinear_wmma_fast(const __bf16* __restrict__ hs,
                                   const __bf16* __restrict__ ts,
                                   const __bf16* __restrict__ Wt,
                                   const float* __restrict__ bias,
                                   float* __restrict__ emb) {
  extern __shared__ char smem[];
  __bf16* ts_sh = (__bf16*)smem;          // 64 x 72
  __bf16* hs_sh = ts_sh + 64 * 72;        // 64 x 8 (2 cols used/stage)
  __bf16* b_sh  = hs_sh + 64 * 8;         // 2 buffers of 64 x 136, [n][k']

  int row0 = blockIdx.x * 64, col0 = blockIdx.y * 64;
  int t = threadIdx.x;
  int wid = t >> 5, lane = t & 31, half = (lane >> 4) & 1, lr = lane & 15;
  int wm = wid >> 2, wn = wid & 3;
  int sc = t >> 2, sq = t & 3;            // W staging map: row sc, quarter sq

  const __bf16* wrow = Wt + (size_t)(col0 + sc) * KTOT_;
  v8f acc0 = {}, acc1 = {};

#ifdef HAVE_ASYNC_LDS
  // prologue: issue slab for stage 0 into buffer 0
  {
    const __bf16* g = wrow;                    // k' base = 0
    __bf16* l = b_sh + sc * 136;
#pragma unroll
    for (int s2 = 0; s2 < 4; s2++)
      async_cp16(g + (sq + 4 * s2) * 8, l + (sq + 4 * s2) * 8);
  }
#endif

  for (int stage = 0; stage < 384; stage++) {  // 384 * 128 = 49152 = K
    int cur = stage & 1;
    int g  = stage >> 5;                       // 32 stages per g-group
    int ib = (stage & 31) * 2;                 // 2 i-columns per stage
    if ((stage & 31) == 0) {
      int j = t & 63, m0 = t >> 6;
#pragma unroll
      for (int s = 0; s < 16; s++) {
        int m = m0 + 4 * s;
        ts_sh[m * 72 + j] = ts[(size_t)(row0 + m) * EMB_ + g * 64 + j];
      }
    }
    if (t < 128) {
      int m = t >> 1, ci = t & 1;
      hs_sh[m * 8 + ci] = hs[(size_t)(row0 + m) * EMB_ + g * 64 + ib + ci];
    }

#ifdef HAVE_ASYNC_LDS
    if (stage + 1 < 384) {                     // issue next slab into alt buf
      const __bf16* gn = wrow + (size_t)(stage + 1) * 128;
      __bf16* ln = b_sh + (cur ^ 1) * (64 * 136) + sc * 136;
#pragma unroll
      for (int s2 = 0; s2 < 4; s2++)
        async_cp16(gn + (sq + 4 * s2) * 8, ln + (sq + 4 * s2) * 8);
      __builtin_amdgcn_s_wait_asynccnt(4);     // drain current slab only
    } else {
      __builtin_amdgcn_s_wait_asynccnt(0);
    }
#else
    {                                          // reg-staged copy, single buffer
      cur = 0;
      const v4u* gr = (const v4u*)(wrow + (size_t)stage * 128);
      v4u tmp[4];
#pragma unroll
      for (int s2 = 0; s2 < 4; s2++) tmp[s2] = gr[sq + 4 * s2];
      v4u* lw = (v4u*)(b_sh + sc * 136);
#pragma unroll
      for (int s2 = 0; s2 < 4; s2++) lw[sq + 4 * s2] = tmp[s2];
    }
#endif
    __syncthreads();

    const v4u* tr0 = (const v4u*)(ts_sh + (wm * 32 + lr) * 72);
    const v4u* tr1 = (const v4u*)(ts_sh + (wm * 32 + 16 + lr) * 72);
    const __bf16* hr0 = hs_sh + (wm * 32 + lr) * 8;
    const __bf16* hr1 = hs_sh + (wm * 32 + 16 + lr) * 8;
    const v4u* br = (const v4u*)(b_sh + cur * (64 * 136) + (wn * 16 + lr) * 136);

#pragma unroll
    for (int s = 0; s < 4; s++) {
      int ii = s >> 1;
      int j0 = (s & 1) * 32;
      Frag ft0, ft1, fb, fa0, fa1;
      int u0 = (j0 + half * 8) >> 3;
      int u1 = u0 + 2;
      ft0.u[0] = tr0[u0]; ft0.u[1] = tr0[u1];
      ft1.u[0] = tr1[u0]; ft1.u[1] = tr1[u1];
      fb.u[0] = br[4 * s + half];
      fb.u[1] = br[4 * s + 2 + half];
      float h0 = (float)hr0[ii];
      float h1 = (float)hr1[ii];
#pragma unroll
      for (int e = 0; e < 16; e++) {
        fa0.v[e] = (__bf16)(h0 * (float)ft0.v[e]);
        fa1.v[e] = (__bf16)(h1 * (float)ft1.v[e]);
      }
      acc0 = __builtin_amdgcn_wmma_f32_16x16x32_bf16(false, fa0.v, false, fb.v,
                                                     (short)0, acc0, false, false);
      acc1 = __builtin_amdgcn_wmma_f32_16x16x32_bf16(false, fa1.v, false, fb.v,
                                                     (short)0, acc1, false, false);
    }
    __syncthreads();
  }

  int col = col0 + wn * 16 + lr;
  float bv = bias[col];
#pragma unroll
  for (int v = 0; v < 8; v++) {
    int r = row0 + wm * 32 + v + 8 * half;
    emb[(size_t)r * EMB_ + col] = acc0[v] + bv;
    emb[(size_t)(r + 16) * EMB_ + col] = acc1[v] + bv;
  }
}

// ============================================================
// K5 (fallback, small ws): f32-W bilinear (round-1 version)
// ============================================================
__global__ void bilinear_wmma(const __bf16* __restrict__ hs,
                              const __bf16* __restrict__ ts,
                              const float* __restrict__ W,
                              const float* __restrict__ bias,
                              float* __restrict__ emb) {
  extern __shared__ char smem[];
  __bf16* ts_sh = (__bf16*)smem;         // 64 x 72
  __bf16* hs_sh = ts_sh + 64 * 72;       // 64 x 8
  __bf16* b_sh  = hs_sh + 64 * 8;        // 64 x 264

  int row0 = blockIdx.x * 64, col0 = blockIdx.y * 64;
  int t = threadIdx.x;
  int wid = t >> 5, lane = t & 31, half = (lane >> 4) & 1, lr = lane & 15;
  int wm = wid >> 2, wn = wid & 3;

  v8f acc0 = {}, acc1 = {};

  for (int stage = 0; stage < 192; stage++) {
    int g  = stage >> 4;
    int ib = (stage & 15) * 4;
    if ((stage & 15) == 0) {
      int j = t & 63, m0 = t >> 6;
#pragma unroll
      for (int s = 0; s < 16; s++) {
        int m = m0 + 4 * s;
        ts_sh[m * 72 + j] = ts[(size_t)(row0 + m) * EMB_ + g * 64 + j];
      }
    }
    {
      int ci = t & 3, m = t >> 2;
      hs_sh[m * 8 + ci] = hs[(size_t)(row0 + m) * EMB_ + g * 64 + ib + ci];
      int c = t & 63, k0 = t >> 6;
      size_t kbase = (size_t)stage * 256;
#pragma unroll 4
      for (int s = 0; s < 64; s++) {
        int k = k0 + 4 * s;
        b_sh[c * 264 + k] = (__bf16)W[(kbase + k) * EMB_ + col0 + c];
      }
    }
    __syncthreads();

    const v4u* tr0 = (const v4u*)(ts_sh + (wm * 32 + lr) * 72);
    const v4u* tr1 = (const v4u*)(ts_sh + (wm * 32 + 16 + lr) * 72);
    const __bf16* hr0 = hs_sh + (wm * 32 + lr) * 8;
    const __bf16* hr1 = hs_sh + (wm * 32 + 16 + lr) * 8;
    const v4u* br = (const v4u*)(b_sh + (wn * 16 + lr) * 264);

#pragma unroll
    for (int s = 0; s < 8; s++) {
      int ii = s >> 1;
      int j0 = (s & 1) * 32;
      Frag ft0, ft1, fb, fa0, fa1;
      int u0 = (j0 + half * 8) >> 3;
      int u1 = u0 + 2;
      ft0.u[0] = tr0[u0]; ft0.u[1] = tr0[u1];
      ft1.u[0] = tr1[u0]; ft1.u[1] = tr1[u1];
      fb.u[0] = br[4 * s + half];
      fb.u[1] = br[4 * s + 2 + half];
      float h0 = (float)hr0[ii];
      float h1 = (float)hr1[ii];
#pragma unroll
      for (int e = 0; e < 16; e++) {
        fa0.v[e] = (__bf16)(h0 * (float)ft0.v[e]);
        fa1.v[e] = (__bf16)(h1 * (float)ft1.v[e]);
      }
      acc0 = __builtin_amdgcn_wmma_f32_16x16x32_bf16(false, fa0.v, false, fb.v,
                                                     (short)0, acc0, false, false);
      acc1 = __builtin_amdgcn_wmma_f32_16x16x32_bf16(false, fa1.v, false, fb.v,
                                                     (short)0, acc1, false, false);
    }
    __syncthreads();
  }

  int col = col0 + wn * 16 + lr;
  float bv = bias[col];
#pragma unroll
  for (int v = 0; v < 8; v++) {
    int r = row0 + wm * 32 + v + 8 * half;
    emb[(size_t)r * EMB_ + col] = acc0[v] + bv;
    emb[(size_t)(r + 16) * EMB_ + col] = acc1[v] + bv;
  }
}

// ============================================================
// K6: r_n = l2norm(r_emb_table @ re_W + re_b)
// ============================================================
__global__ void rel_emb(const float* __restrict__ r_tab,
                        const float* __restrict__ re_W,
                        const float* __restrict__ re_b,
                        float* __restrict__ r_n) {
  __shared__ float rrow[EMB_];
  __shared__ float outr[EMB_];
  __shared__ float red[256];
  int i = blockIdx.x;
  for (int d = threadIdx.x; d < EMB_; d += blockDim.x)
    rrow[d] = r_tab[(size_t)i * EMB_ + d];
  __syncthreads();
  float lsum = 0.f;
  for (int c = threadIdx.x; c < EMB_; c += blockDim.x) {
    float acc = re_b[c];
    for (int d = 0; d < EMB_; d++) acc += rrow[d] * re_W[(size_t)d * EMB_ + c];
    outr[c] = acc;
    lsum += acc * acc;
  }
  float S = block_reduce_sum(lsum, red);
  float inv = 1.0f / fmaxf(sqrtf(S), 1e-12f);
  for (int c = threadIdx.x; c < EMB_; c += blockDim.x)
    r_n[(size_t)i * EMB_ + c] = outr[c] * inv;
}

// ============================================================
// K7: logits = l2norm(emb) @ r_n^T
// ============================================================
__global__ void logits_k(const float* __restrict__ emb,
                         const float* __restrict__ r_n,
                         float* __restrict__ out) {
  __shared__ float row[EMB_];
  __shared__ float red[128];
  int r = blockIdx.x;
  float lsq = 0.f;
  for (int d = threadIdx.x; d < EMB_; d += blockDim.x) {
    float v = emb[(size_t)r * EMB_ + d];
    row[d] = v;
    lsq += v * v;
  }
  float S = block_reduce_sum(lsq, red);
  float inv = 1.0f / fmaxf(sqrtf(S), 1e-12f);
  for (int c = threadIdx.x; c < NC_; c += blockDim.x) {
    float acc = 0.f;
    for (int d = 0; d < EMB_; d++) acc += row[d] * r_n[(size_t)c * EMB_ + d];
    out[(size_t)r * NC_ + c] = acc * inv;
  }
}

// ============================================================
extern "C" void kernel_launch(void* const* d_in, const int* in_sizes, int n_in,
                              void* d_out, int out_size, void* d_ws, size_t ws_size,
                              hipStream_t stream) {
  (void)in_sizes; (void)n_in; (void)out_size;
  const float* seq    = (const float*)d_in[0];
  const float* att    = (const float*)d_in[1];
  const int*   m_idx  = (const int*)d_in[2];
  const unsigned char* m_mask = (const unsigned char*)d_in[3];
  const int*   pairs  = (const int*)d_in[4];
  const float* head_W = (const float*)d_in[5];
  const float* head_b = (const float*)d_in[6];
  const float* tail_W = (const float*)d_in[7];
  const float* tail_b = (const float*)d_in[8];
  const float* bil_W  = (const float*)d_in[9];
  const float* bil_b  = (const float*)d_in[10];
  const float* r_tab  = (const float*)d_in[11];
  const float* re_W   = (const float*)d_in[12];
  const float* re_b   = (const float*)d_in[13];
  float* out = (float*)d_out;

  char* ws = (char*)d_ws;
  float*  e_emb = (float*)ws;  ws += 393216;     // 4*32*768 f32
  float*  e_att = (float*)ws;  ws += 6291456;    // 4*32*12*1024 f32
  float*  ht    = (float*)ws;  ws += 4194304;    // 4*256*1024 f32
  float*  catH  = (float*)ws;  ws += 6291456;    // 1024*1536 f32
  float*  catT  = (float*)ws;  ws += 6291456;
  __bf16* hs_bf = (__bf16*)ws; ws += 1572864;    // 1024*768 bf16
  __bf16* ts_bf = (__bf16*)ws; ws += 1572864;
  float*  emb   = (float*)ws;  ws += 3145728;    // 1024*768 f32
  float*  r_n   = (float*)ws;  ws += 297984;     // 97*768 f32
  __bf16* Wt    = (__bf16*)ws; ws += 75497472;   // 768*49152 bf16
  const size_t NEED_BIG = 30051328ull + 75497472ull;  // 105548800
  bool big = ws_size >= NEED_BIG;

  entity_agg<<<dim3(N_ * E_), dim3(256), 0, stream>>>(seq, att, m_idx, m_mask,
                                                      e_emb, e_att);
  pair_ht<<<dim3(N_ * P_), dim3(256), 0, stream>>>(e_att, pairs, ht);
  gather_pairs<<<dim3((N_ * P_ * D_ + 255) / 256), dim3(256), 0, stream>>>(
      e_emb, pairs, catH, catT);
  gemm_wmma<false, false><<<dim3(4, 12, 4), dim3(256), 10240, stream>>>(
      ht, seq, nullptr, (void*)(catH + D_), (void*)(catT + D_),
      P_, D_, C_, C_, D_, 2 * D_,
      (long)P_ * C_, (long)C_ * D_, (long)P_ * 2 * D_);
  gemm_wmma<true, true><<<dim3(16, 12, 1), dim3(256), 10240, stream>>>(
      catH, head_W, head_b, (void*)hs_bf, nullptr,
      N_ * P_, EMB_, 2 * D_, 2 * D_, EMB_, EMB_, 0L, 0L, 0L);
  gemm_wmma<true, true><<<dim3(16, 12, 1), dim3(256), 10240, stream>>>(
      catT, tail_W, tail_b, (void*)ts_bf, nullptr,
      N_ * P_, EMB_, 2 * D_, 2 * D_, EMB_, EMB_, 0L, 0L, 0L);
  if (big) {
    conv_transpose_W<<<dim3(KTOT_ / 64, EMB_ / 64), dim3(256), 0, stream>>>(
        bil_W, Wt);
    bilinear_wmma_fast<<<dim3(16, 12), dim3(256), 45056, stream>>>(
        hs_bf, ts_bf, Wt, bil_b, emb);
  } else {
    bilinear_wmma<<<dim3(16, 12), dim3(256), 44032, stream>>>(
        hs_bf, ts_bf, bil_W, bil_b, emb);
  }
  rel_emb<<<dim3(NC_), dim3(256), 0, stream>>>(r_tab, re_W, re_b, r_n);
  logits_k<<<dim3(N_ * P_), dim3(128), 0, stream>>>(emb, r_n, out);
}
